// VQVAE_62208306315484
// MI455X (gfx1250) — matrix operand order repaired and checked by reading, not compiled, not run on hardware
//
#include <hip/hip_runtime.h>
#include <math.h>

typedef __bf16 bf16;
typedef __attribute__((ext_vector_type(16))) __bf16 v16bf;
typedef __attribute__((ext_vector_type(8)))  __bf16 v8bf;
typedef __attribute__((ext_vector_type(8)))  float  v8f;

static __device__ __forceinline__ v8bf ld8(const bf16* p) { return *(const v8bf*)p; }
static __device__ __forceinline__ v16bf cat16(v8bf a, v8bf b) {
  return __builtin_shufflevector(a, b, 0,1,2,3,4,5,6,7,8,9,10,11,12,13,14,15);
}

static inline int ngrid(long n, int t, int cap = 8192) {
  long b = (n + t - 1) / t;
  if (b > cap) b = cap;
  if (b < 1) b = 1;
  return (int)b;
}
static inline int ilog2i(int v) { int l = 0; while ((1 << l) < v) ++l; return l; }

// ---------------------------------------------------------------------------
// fp32 -> bf16 cast (codebook)
// ---------------------------------------------------------------------------
__global__ void k_cast_f2b(const float* __restrict__ s, bf16* __restrict__ d, long n) {
  long i  = (long)blockIdx.x * blockDim.x + threadIdx.x;
  long st = (long)gridDim.x * blockDim.x;
  for (; i < n; i += st) d[i] = (bf16)s[i];
}

// ---------------------------------------------------------------------------
// pack conv weights OIHW fp32 -> [Coutp, (r*kw+s)*Cinp + ci] bf16, zero padded.
// K-order (rs, ci) makes im2col decomposition shift/mask for pow2 Cinp.
// ---------------------------------------------------------------------------
__global__ void k_pack_w(const float* __restrict__ src, bf16* __restrict__ dst,
                         int Cout, int Coutp, int Cin, int Cinp, int kw) {
  int  K  = kw * kw * Cinp;
  long n  = (long)Coutp * K;
  long i  = (long)blockIdx.x * blockDim.x + threadIdx.x;
  long st = (long)gridDim.x * blockDim.x;
  for (; i < n; i += st) {
    int m  = (int)(i / K);
    int k  = (int)(i - (long)m * K);
    int rs = k / Cinp;
    int ci = k - rs * Cinp;
    int r  = rs / kw, s = rs - r * kw;
    float v = (m < Cout && ci < Cin) ? src[(((long)m * Cin + ci) * kw + r) * kw + s] : 0.f;
    dst[i] = (bf16)v;
  }
}

// pack input image NCHW fp32 (C=3) -> NHWC bf16 (Cp=4, ch3 = 0)
__global__ void k_pack_in(const float* __restrict__ x, bf16* __restrict__ y,
                          int B, int C, int Cp, int HW) {
  long n  = (long)B * HW * Cp;
  long i  = (long)blockIdx.x * blockDim.x + threadIdx.x;
  long st = (long)gridDim.x * blockDim.x;
  for (; i < n; i += st) {
    int  c = (int)(i & (Cp - 1));
    long t = i >> 2;                       // Cp == 4
    int  p = (int)(t % HW);
    int  b = (int)(t / HW);
    y[i] = (bf16)((c < C) ? x[((long)b * C + c) * HW + p] : 0.f);
  }
}

// ---------------------------------------------------------------------------
// GroupNorm (NHWC): stats per (batch, group), then apply (+optional SiLU)
// ---------------------------------------------------------------------------
__global__ void k_gn_stats(const bf16* __restrict__ x, float* __restrict__ stats,
                           int HW, int C, int cgLog2) {
  int bg = blockIdx.x;                     // b*32 + g
  int b  = bg >> 5;
  int g  = bg & 31;
  int Cg = 1 << cgLog2;
  long base = (long)b * HW * C + (long)g * Cg;
  long n    = (long)HW << cgLog2;
  float s = 0.f, s2 = 0.f;
  for (long i = threadIdx.x; i < n; i += blockDim.x) {
    long p  = i >> cgLog2;
    int  cc = (int)(i & (Cg - 1));
    float v = (float)x[base + p * C + cc];
    s += v; s2 += v * v;
  }
  __shared__ float sh[256], sh2[256];
  sh[threadIdx.x] = s; sh2[threadIdx.x] = s2;
  __syncthreads();
  for (int o = blockDim.x / 2; o > 0; o >>= 1) {
    if ((int)threadIdx.x < o) { sh[threadIdx.x] += sh[threadIdx.x + o]; sh2[threadIdx.x] += sh2[threadIdx.x + o]; }
    __syncthreads();
  }
  if (threadIdx.x == 0) {
    float m = sh[0] / (float)n;
    float v = sh2[0] / (float)n - m * m;
    stats[bg * 2 + 0] = m;
    stats[bg * 2 + 1] = v;
  }
}

__global__ void k_gn_apply(const bf16* __restrict__ x, bf16* __restrict__ y,
                           const float* __restrict__ stats,
                           const float* __restrict__ gamma, const float* __restrict__ beta,
                           long n, int cLog2, int hwLog2, int cgLog2, int dosilu) {
  long i  = (long)blockIdx.x * blockDim.x + threadIdx.x;
  long st = (long)gridDim.x * blockDim.x;
  int  C  = 1 << cLog2;
  for (; i < n; i += st) {
    int c  = (int)(i & (C - 1));
    int b  = (int)(i >> (cLog2 + hwLog2));
    int bg = (b << 5) + (c >> cgLog2);
    float m = stats[bg * 2 + 0], v = stats[bg * 2 + 1];
    float val = ((float)x[i] - m) * rsqrtf(v + 1e-6f) * gamma[c] + beta[c];
    if (dosilu) val = val / (1.0f + __expf(-val));
    y[i] = (bf16)val;
  }
}

// ---------------------------------------------------------------------------
// Implicit-GEMM conv, NHWC, WMMA bf16 16x16x32.  Wave = 16(cout) x 16(pixel).
// VEC: Cinp>=16 pow2 -> all operand loads are 16B vector loads, one bounds
// check per 16-element im2col run.  Epilogue: fused bias(+residual), 16B
// NHWC bf16 store, or scalar NCHW fp32 store for the network output.
// ---------------------------------------------------------------------------
template<int KW, int STRIDE, int PAD, bool VEC>
__global__ void __launch_bounds__(32)
k_conv(const bf16* __restrict__ X, const bf16* __restrict__ Wt,
       const float* __restrict__ bias, const bf16* __restrict__ res,
       bf16* __restrict__ Y, float* __restrict__ Yf,
       int B, int cinLog2, int H, int Wd,
       int Cout, int owSh, int ohowSh) {
  const int lane = threadIdx.x;
  const int half = lane >> 4;
  const int Cinp = 1 << cinLog2;
  const int K    = KW * KW * Cinp;
  const int p0   = blockIdx.x * 16;
  const int m0   = blockIdx.y * 16;
  const int P    = B << ohowSh;

  int  pn = p0 + (lane & 15);
  bool pv = pn < P;
  int  pp = pv ? pn : 0;
  int  bb  = pp >> ohowSh;
  int  rem = pp & ((1 << ohowSh) - 1);
  int  oh  = rem >> owSh;
  int  ow  = rem & ((1 << owSh) - 1);
  int  mrow = m0 + (lane & 15);
  const bf16* wrow = Wt + (long)mrow * K;

  v8f acc = {};
  for (int k0 = 0; k0 < K; k0 += 32) {
    __builtin_prefetch(wrow + k0 + 32, 0, 1);
    v16bf fa, fb;
    if (VEC) {
      // A: weights, rows padded to 16 -> unguarded 16B loads
      const bf16* wp = wrow + k0 + (half ? 16 : 0);
      fa = cat16(ld8(wp), ld8(wp + 8));
      // B: one (r,s) tap, 16 contiguous input channels
      int kbB = k0 + (half ? 16 : 0);
      int rs  = kbB >> cinLog2;
      int ci0 = kbB & (Cinp - 1);
      int r   = rs / KW, s = rs - r * KW;
      int ih  = oh * STRIDE - PAD + r;
      int iw  = ow * STRIDE - PAD + s;
      v8bf b0 = {}, b1 = {};
      if (pv && ih >= 0 && ih < H && iw >= 0 && iw < Wd) {
        const bf16* xp = X + ((((long)bb * H + ih) * Wd + iw) << cinLog2) + ci0;
        b0 = ld8(xp); b1 = ld8(xp + 8);
      }
      fb = cat16(b0, b1);
    } else {
      // scalar guarded path (conv_in: Cinp=4, K=36)
#pragma unroll
      for (int j = 0; j < 16; ++j) {
        int kk = k0 + (half ? 16 : 0) + j;
        fa[j] = (kk < K) ? wrow[kk] : (bf16)0.0f;
      }
#pragma unroll
      for (int j = 0; j < 16; ++j) {
        int kk = k0 + (half ? 16 : 0) + j;
        bf16 v = (bf16)0.0f;
        if (pv && kk < K) {
          int rs = kk >> cinLog2;
          int ci = kk & (Cinp - 1);
          int r  = rs / KW, s = rs - r * KW;
          int ih = oh * STRIDE - PAD + r;
          int iw = ow * STRIDE - PAD + s;
          if (ih >= 0 && ih < H && iw >= 0 && iw < Wd)
            v = X[((((long)bb * H + ih) * Wd + iw) << cinLog2) + ci];
        }
        fb[j] = v;
      }
    }
    acc = __builtin_amdgcn_wmma_f32_16x16x32_bf16(false, fa, false, fb, (short)0, acc, false, false);
  }
  if (!pv) return;
  int mb = m0 + (half ? 8 : 0);
  if (Yf) {
    // network output: fp32 NCHW, Cout may be < 16 (rows zero-padded in Wt)
    int OWv = 1 << owSh;
    int OHv = 1 << (ohowSh - owSh);
#pragma unroll
    for (int r = 0; r < 8; ++r) {
      int m = mb + r;
      if (m < Cout)
        Yf[(((long)bb * Cout + m) * OHv + oh) * OWv + ow] = acc[r] + bias[m];
    }
  } else {
    long off = (long)pn * Cout + mb;
    v8bf rv = {};
    if (res) rv = ld8(res + off);
    v8bf ov;
#pragma unroll
    for (int r = 0; r < 8; ++r) {
      float val = acc[r] + bias[mb + r];
      if (res) val += (float)rv[r];
      ov[r] = (bf16)val;
    }
    *(v8bf*)(Y + off) = ov;
  }
}

// ---------------------------------------------------------------------------
// Generic WMMA GEMM, bf16 in / fp32 acc.  A[m,k]=A[m*sAm+k*sAk],
// B[k,n]=B[k*sBk+n*sBn].  AV/BV: contiguous-K vector-load fast paths
// (requires sAk==1 / sBk==1, M,N mult 16, K mult 32).
// ---------------------------------------------------------------------------
template<bool AV, bool BV>
__global__ void __launch_bounds__(32)
k_gemm(const bf16* __restrict__ A, long sAm, long sAk,
       const bf16* __restrict__ Bm, long sBk, long sBn,
       float alpha,
       bf16* __restrict__ D, float* __restrict__ Df,
       long sDm, long sDn, int M, int N, int K) {
  int lane = threadIdx.x, half = lane >> 4;
  int n0 = blockIdx.x * 16, m0 = blockIdx.y * 16;
  int mrow = m0 + (lane & 15);
  int ncol = n0 + (lane & 15);
  v8f acc = {};
  for (int k0 = 0; k0 < K; k0 += 32) {
    v16bf fa, fb;
    if (AV) {
      const bf16* ap = A + (long)mrow * sAm + k0 + (half ? 16 : 0);
      fa = cat16(ld8(ap), ld8(ap + 8));
    } else {
#pragma unroll
      for (int j = 0; j < 16; ++j) {
        int kk = k0 + (half ? 16 : 0) + j;
        fa[j] = A[(long)mrow * sAm + (long)kk * sAk];
      }
    }
    if (BV) {
      const bf16* bp = Bm + (long)ncol * sBn + k0 + (half ? 16 : 0);
      fb = cat16(ld8(bp), ld8(bp + 8));
    } else {
#pragma unroll
      for (int j = 0; j < 16; ++j) {
        int kk = k0 + (half ? 16 : 0) + j;
        fb[j] = Bm[(long)kk * sBk + (long)ncol * sBn];
      }
    }
    acc = __builtin_amdgcn_wmma_f32_16x16x32_bf16(false, fa, false, fb, (short)0, acc, false, false);
  }
  int mb = m0 + (half ? 8 : 0);
  if (Df) {
#pragma unroll
    for (int r = 0; r < 8; ++r)
      Df[(long)(mb + r) * sDm + (long)ncol * sDn] = acc[r] * alpha;
  } else if (sDm == 1) {
    v8bf ov;
#pragma unroll
    for (int r = 0; r < 8; ++r) ov[r] = (bf16)(acc[r] * alpha);
    *(v8bf*)(D + (long)ncol * sDn + mb) = ov;
  } else {
#pragma unroll
    for (int r = 0; r < 8; ++r)
      D[(long)(mb + r) * sDm + (long)ncol * sDn] = (bf16)(acc[r] * alpha);
  }
}

// ---------------------------------------------------------------------------
// Row softmax: fp32 scores -> bf16 probs.  blockDim == 256.
// ---------------------------------------------------------------------------
__global__ void k_softmax_row(const float* __restrict__ S, bf16* __restrict__ P, int N) {
  int row = blockIdx.x;
  const float* s = S + (long)row * N;
  bf16*        p = P + (long)row * N;
  __shared__ float sh[256];
  float mx = -3.0e38f;
  for (int i = threadIdx.x; i < N; i += 256) mx = fmaxf(mx, s[i]);
  sh[threadIdx.x] = mx; __syncthreads();
  for (int o = 128; o > 0; o >>= 1) {
    if ((int)threadIdx.x < o) sh[threadIdx.x] = fmaxf(sh[threadIdx.x], sh[threadIdx.x + o]);
    __syncthreads();
  }
  mx = sh[0]; __syncthreads();
  float sum = 0.f;
  for (int i = threadIdx.x; i < N; i += 256) sum += __expf(s[i] - mx);
  sh[threadIdx.x] = sum; __syncthreads();
  for (int o = 128; o > 0; o >>= 1) {
    if ((int)threadIdx.x < o) sh[threadIdx.x] += sh[threadIdx.x + o];
    __syncthreads();
  }
  float inv = 1.0f / sh[0];
  for (int i = threadIdx.x; i < N; i += 256) p[i] = (bf16)(__expf(s[i] - mx) * inv);
}

// ---------------------------------------------------------------------------
// nearest-neighbor 2x upsample (NHWC bf16), pow2 dims -> shifts only
// ---------------------------------------------------------------------------
__global__ void k_up2x(const bf16* __restrict__ x, bf16* __restrict__ y,
                       long n, int cLog2, int hLog2, int wLog2) {
  long i  = (long)blockIdx.x * blockDim.x + threadIdx.x;
  long st = (long)gridDim.x * blockDim.x;
  int  C  = 1 << cLog2;
  int  H  = 1 << hLog2, W = 1 << wLog2;
  for (; i < n; i += st) {
    int  c  = (int)(i & (C - 1));
    long t  = i >> cLog2;
    int  ow = (int)(t & (2 * W - 1)); t >>= (wLog2 + 1);
    int  oh = (int)(t & (2 * H - 1)); t >>= (hLog2 + 1);   // t = b
    y[i] = x[((((t << hLog2) + (oh >> 1)) << wLog2) + (ow >> 1) << cLog2) + c];
  }
}

// ---------------------------------------------------------------------------
// codebook squared-norms
// ---------------------------------------------------------------------------
__global__ void k_cbnorm(const bf16* __restrict__ cb, float* __restrict__ cbn, int NE, int C) {
  int i = blockIdx.x * blockDim.x + threadIdx.x;
  if (i >= NE) return;
  float s = 0.f;
  for (int c = 0; c < C; ++c) { float v = (float)cb[(long)i * C + c]; s += v * v; }
  cbn[i] = s;
}

// ---------------------------------------------------------------------------
// fused VQ argmin: z NHWC rows are contiguous -> fully vectorized WMMA
// streaming over all codes, running (dist,idx) in registers.
// ---------------------------------------------------------------------------
#define VQ_C 256
__global__ void __launch_bounds__(32)
k_vq_argmin(const bf16* __restrict__ Z, const bf16* __restrict__ CB,
            const float* __restrict__ cbn, int* __restrict__ idxO, int NE) {
  int lane = threadIdx.x, half = lane >> 4;
  int pn = blockIdx.x * 16 + (lane & 15);
  const bf16* zp = Z + (long)pn * VQ_C;
  v16bf bz[VQ_C / 32];
#pragma unroll
  for (int ks = 0; ks < VQ_C / 32; ++ks) {
    const bf16* p = zp + ks * 32 + (half ? 16 : 0);
    bz[ks] = cat16(ld8(p), ld8(p + 8));
  }
  float best = 3.4e38f; int bidx = 0;
  for (int c0 = 0; c0 < NE; c0 += 16) {
    const bf16* crow = CB + (long)(c0 + (lane & 15)) * VQ_C;
    v8f acc = {};
#pragma unroll
    for (int ks = 0; ks < VQ_C / 32; ++ks) {
      const bf16* p = crow + ks * 32 + (half ? 16 : 0);
      v16bf fa = cat16(ld8(p), ld8(p + 8));
      acc = __builtin_amdgcn_wmma_f32_16x16x32_bf16(false, fa, false, bz[ks], (short)0, acc, false, false);
    }
#pragma unroll
    for (int r = 0; r < 8; ++r) {
      int   code = c0 + r + (half ? 8 : 0);
      float d    = cbn[code] - 2.0f * acc[r];
      if (d < best || (d == best && code < bidx)) { best = d; bidx = code; }
    }
  }
  float ob = __shfl_xor(best, 16);
  int   oi = __shfl_xor(bidx, 16);
  if (ob < best || (ob == best && oi < bidx)) { best = ob; bidx = oi; }
  if (lane < 16) idxO[pn] = bidx;
}

// zq[p, 0:256] = cb[idx[p], 0:256], 8 channels per thread (16B copies)
__global__ void k_vq_gather(const int* __restrict__ idx, const bf16* __restrict__ cb,
                            bf16* __restrict__ out, long P) {
  long n  = P * (VQ_C / 8);
  long i  = (long)blockIdx.x * blockDim.x + threadIdx.x;
  long st = (long)gridDim.x * blockDim.x;
  for (; i < n; i += st) {
    long p  = i >> 5;                         // VQ_C/8 == 32
    int  c8 = (int)(i & 31) * 8;
    *(v8bf*)(out + p * VQ_C + c8) = ld8(cb + (long)idx[p] * VQ_C + c8);
  }
}

// ---------------------------------------------------------------------------
// host orchestration
// ---------------------------------------------------------------------------
extern "C" void kernel_launch(void* const* d_in, const int* in_sizes, int n_in,
                              void* d_out, int out_size, void* d_ws, size_t ws_size,
                              hipStream_t stream) {
  (void)in_sizes; (void)n_in; (void)out_size; (void)ws_size;
  int ci = 0;
  auto F = [&]() { return (const float*)d_in[ci++]; };

  struct ConvP { const float* b; const float* w; int cin, cout, k, cinp, cinLog2, coutp; bf16* wbf; };
  struct NormP { const float* be; const float* g; };
  struct ResP  { ConvP c1, c2, nin; NormP n1, n2; int has_nin; };
  struct AttnP { ConvP kk, proj, q, v; NormP norm; };
  struct MidP  { AttnP attn; ResP b1, b2; };

  auto readConv = [&](int cin, int cout, int k) {
    ConvP p; p.b = F(); p.w = F(); p.cin = cin; p.cout = cout; p.k = k;
    p.cinp = (cin < 16) ? 4 : cin;
    p.cinLog2 = ilog2i(p.cinp);
    p.coutp = (cout + 15) & ~15;
    p.wbf = nullptr;
    return p;
  };
  auto readNorm = [&]() { NormP p; p.be = F(); p.g = F(); return p; };
  auto readRes  = [&](int cin, int cout) {
    ResP r; r.c1 = readConv(cin, cout, 3); r.c2 = readConv(cout, cout, 3);
    r.has_nin = (cin != cout);
    if (r.has_nin) r.nin = readConv(cin, cout, 1);
    r.n1 = readNorm(); r.n2 = readNorm();
    return r;
  };
  auto readAttn = [&](int c) {
    AttnP a; a.kk = readConv(c, c, 1); a.norm = readNorm(); a.proj = readConv(c, c, 1);
    a.q = readConv(c, c, 1); a.v = readConv(c, c, 1); return a;
  };
  auto readMid = [&](int c) { MidP m; m.attn = readAttn(c); m.b1 = readRes(c, c); m.b2 = readRes(c, c); return m; };

  // ---- parse in JAX pytree flatten order (sorted dict keys, list order) ----
  ConvP dec_conv_in  = readConv(256, 512, 3);
  ConvP dec_conv_out = readConv(128,   3, 3);
  MidP  dec_mid      = readMid(512);
  NormP dec_norm_out = readNorm();
  ResP  dU0b[3] = { readRes(256, 128), readRes(128, 128), readRes(128, 128) };
  ResP  dU1b[3] = { readRes(512, 256), readRes(256, 256), readRes(256, 256) };
  ConvP dU1up   = readConv(256, 256, 3);
  ResP  dU2b[3] = { readRes(512, 512), readRes(512, 512), readRes(512, 512) };
  ConvP dU2up   = readConv(512, 512, 3);

  ConvP enc_conv_in  = readConv(  3, 128, 3);
  ConvP enc_conv_out = readConv(512, 256, 3);
  ResP  eD0b[2] = { readRes(128, 128), readRes(128, 128) };
  ConvP eD0dn   = readConv(128, 128, 3);
  ResP  eD1b[2] = { readRes(128, 256), readRes(256, 256) };
  ConvP eD1dn   = readConv(256, 256, 3);
  ResP  eD2b[2] = { readRes(256, 512), readRes(512, 512) };
  MidP  enc_mid      = readMid(512);
  NormP enc_norm_out = readNorm();

  ConvP post_quant = readConv(256, 256, 1);
  ConvP quant_conv = readConv(256, 256, 1);
  const float* codebook = F();
  const float* x_in     = F();

  // ---- workspace arena -----------------------------------------------------
  size_t off = 0;
  auto alloc = [&](size_t bytes) {
    void* p = (char*)d_ws + off;
    off += (bytes + 255) & ~(size_t)255;
    return p;
  };
  const long MAXE = 2L * 256 * 256 * 256;
  bf16* Bb[6];
  for (int i = 0; i < 6; ++i) Bb[i] = (bf16*)alloc(MAXE * sizeof(bf16));
  float* scores = (float*)alloc(4096L * 4096 * sizeof(float));
  bf16*  probs  = (bf16*) alloc(4096L * 4096 * sizeof(bf16));
  float* stats  = (float*)alloc(64 * 2 * sizeof(float));
  bf16*  cbbf   = (bf16*) alloc(8192L * 256 * sizeof(bf16));
  float* cbn    = (float*)alloc(8192 * sizeof(float));
  int*   vidx   = (int*)  alloc(8192 * sizeof(int));

  // pack all conv weights -> bf16 (rs,ci)-ordered, padded rows/channels
  auto prepConv = [&](ConvP& p) {
    long n = (long)p.coutp * p.k * p.k * p.cinp;
    p.wbf = (bf16*)alloc(n * sizeof(bf16));
    k_pack_w<<<ngrid(n, 256), 256, 0, stream>>>(p.w, p.wbf, p.cout, p.coutp, p.cin, p.cinp, p.k);
  };
  auto prepRes  = [&](ResP& r)  { prepConv(r.c1); prepConv(r.c2); if (r.has_nin) prepConv(r.nin); };
  auto prepAttn = [&](AttnP& a) { prepConv(a.kk); prepConv(a.q); prepConv(a.v); prepConv(a.proj); };
  auto prepMid  = [&](MidP& m)  { prepAttn(m.attn); prepRes(m.b1); prepRes(m.b2); };

  prepConv(dec_conv_in); prepConv(dec_conv_out); prepMid(dec_mid);
  for (int i = 0; i < 3; ++i) { prepRes(dU0b[i]); prepRes(dU1b[i]); prepRes(dU2b[i]); }
  prepConv(dU1up); prepConv(dU2up);
  prepConv(enc_conv_in); prepConv(enc_conv_out);
  for (int i = 0; i < 2; ++i) { prepRes(eD0b[i]); prepRes(eD1b[i]); prepRes(eD2b[i]); }
  prepConv(eD0dn); prepConv(eD1dn); prepMid(enc_mid);
  prepConv(post_quant); prepConv(quant_conv);
  k_cast_f2b<<<ngrid(8192L * 256, 256), 256, 0, stream>>>(codebook, cbbf, 8192L * 256);
  k_cbnorm<<<(8192 + 255) / 256, 256, 0, stream>>>(cbbf, cbn, 8192, 256);

  // ---- forward (activations: NHWC bf16) -----------------------------------
  struct T { bf16* p; int B, C, H, W; };

  auto runConv = [&](T in, ConvP& cp, int stride, const bf16* res,
                     bf16* out, float* outf, int OH, int OW) {
    int P = in.B * OH * OW;
    int owSh = ilog2i(OW), ohowSh = ilog2i(OH * OW);
    dim3 g((P + 15) / 16, cp.coutp / 16);
    if (cp.k == 3 && stride == 1 && cp.cinp >= 16)
      k_conv<3,1,1,true ><<<g, 32, 0, stream>>>(in.p, cp.wbf, cp.b, res, out, outf,
          in.B, cp.cinLog2, in.H, in.W, cp.cout, owSh, ohowSh);
    else if (cp.k == 3 && stride == 1)
      k_conv<3,1,1,false><<<g, 32, 0, stream>>>(in.p, cp.wbf, cp.b, res, out, outf,
          in.B, cp.cinLog2, in.H, in.W, cp.cout, owSh, ohowSh);
    else if (cp.k == 3)   // stride 2, torch pad (0,1,0,1) == bounds-checked pad 0
      k_conv<3,2,0,true ><<<g, 32, 0, stream>>>(in.p, cp.wbf, cp.b, res, out, outf,
          in.B, cp.cinLog2, in.H, in.W, cp.cout, owSh, ohowSh);
    else
      k_conv<1,1,0,true ><<<g, 32, 0, stream>>>(in.p, cp.wbf, cp.b, res, out, outf,
          in.B, cp.cinLog2, in.H, in.W, cp.cout, owSh, ohowSh);
    T r; r.p = out; r.B = in.B; r.C = cp.cout; r.H = OH; r.W = OW; return r;
  };
  auto runGN = [&](T in, NormP& np, int silu, bf16* out) {
    int cLog2 = ilog2i(in.C), hwLog2 = ilog2i(in.H * in.W), cgLog2 = cLog2 - 5;
    k_gn_stats<<<in.B * 32, 256, 0, stream>>>(in.p, stats, in.H * in.W, in.C, cgLog2);
    long n = (long)in.B * in.C * in.H * in.W;
    k_gn_apply<<<ngrid(n, 256), 256, 0, stream>>>(in.p, out, stats, np.g, np.be,
                                                  n, cLog2, hwLog2, cgLog2, silu);
    T r = in; r.p = out; return r;
  };
  auto runRes = [&](T x, ResP& r, bf16* s1, bf16* s2, bf16* s3, bf16* out) {
    T t  = runGN(x, r.n1, 1, s1);
    T h1 = runConv(t, r.c1, 1, nullptr, s2, nullptr, x.H, x.W);
    t = runGN(h1, r.n2, 1, s1);
    const bf16* resid = x.p;
    if (r.has_nin) { T rr = runConv(x, r.nin, 1, nullptr, s3, nullptr, x.H, x.W); resid = rr.p; }
    return runConv(t, r.c2, 1, resid, out, nullptr, x.H, x.W);
  };
  auto runAttn = [&](T x, AttnP& a, bf16* s1, bf16* s2, bf16* s3, bf16* s4, bf16* out) {
    int HW = x.H * x.W;                                  // 4096
    T hn = runGN(x, a.norm, 0, s1);
    runConv(hn, a.q,  1, nullptr, s2, nullptr, x.H, x.W);
    runConv(hn, a.kk, 1, nullptr, s3, nullptr, x.H, x.W);
    runConv(hn, a.v,  1, nullptr, s4, nullptr, x.H, x.W);
    float scale = 1.0f / sqrtf((float)x.C);
    for (int b = 0; b < x.B; ++b) {
      const bf16* qb = s2 + (long)b * HW * x.C;          // [HW, C]
      const bf16* kb = s3 + (long)b * HW * x.C;
      const bf16* vb = s4 + (long)b * HW * x.C;
      // scores[i,j] = scale * sum_c q[i,c] k[j,c]   (both K-contiguous)
      dim3 g1(HW / 16, HW / 16);
      k_gemm<true,true><<<g1, 32, 0, stream>>>(qb, x.C, 1, kb, 1, x.C, scale,
                                               nullptr, scores, HW, 1, HW, HW, x.C);
      k_softmax_row<<<HW, 256, 0, stream>>>(scores, probs, HW);
      // O[i,c] = sum_j p[i,j] v[j,c];  computed as M=c, N=i (B=probs contiguous,
      // D stores contiguous 8-channel runs)
      dim3 g2(HW / 16, x.C / 16);
      k_gemm<false,true><<<g2, 32, 0, stream>>>(vb, 1, x.C, probs, 1, HW, 1.0f,
                                                s1 + (long)b * HW * x.C, nullptr,
                                                1, x.C, x.C, HW, HW);
    }
    T hb; hb.p = s1; hb.B = x.B; hb.C = x.C; hb.H = x.H; hb.W = x.W;
    return runConv(hb, a.proj, 1, x.p, out, nullptr, x.H, x.W);
  };

  T h; int curb = 1;
  auto other = [&]() { return Bb[curb ^ 1]; };
  auto flip  = [&]() { curb ^= 1; };
  auto res   = [&](ResP& r)  { h = runRes (h, r, Bb[2], Bb[3], Bb[4], other()); flip(); };
  auto attn  = [&](AttnP& a) { h = runAttn(h, a, Bb[2], Bb[3], Bb[4], Bb[5], other()); flip(); };
  auto down  = [&](ConvP& c) { h = runConv(h, c, 2, nullptr, other(), nullptr, h.H / 2, h.W / 2); flip(); };
  auto up    = [&](ConvP& c) {
    long n = (long)h.B * h.C * 4 * h.H * h.W;
    k_up2x<<<ngrid(n, 256), 256, 0, stream>>>(h.p, Bb[2], n, ilog2i(h.C), ilog2i(h.H), ilog2i(h.W));
    T hu; hu.p = Bb[2]; hu.B = h.B; hu.C = h.C; hu.H = h.H * 2; hu.W = h.W * 2;
    h = runConv(hu, c, 1, nullptr, other(), nullptr, hu.H, hu.W); flip();
  };

  // ---- encoder ----
  k_pack_in<<<ngrid(2L * 65536 * 4, 256), 256, 0, stream>>>(x_in, Bb[5], 2, 3, 4, 65536);
  T xi; xi.p = Bb[5]; xi.B = 2; xi.C = 3; xi.H = 256; xi.W = 256;
  h = runConv(xi, enc_conv_in, 1, nullptr, Bb[1], nullptr, 256, 256); curb = 1;
  res(eD0b[0]); res(eD0b[1]); down(eD0dn);
  res(eD1b[0]); res(eD1b[1]); down(eD1dn);
  res(eD2b[0]); res(eD2b[1]);
  res(enc_mid.b1); attn(enc_mid.attn); res(enc_mid.b2);
  h = runGN(h, enc_norm_out, 1, Bb[2]);
  h = runConv(h, enc_conv_out, 1, nullptr, other(), nullptr, h.H, h.W); flip();
  h = runConv(h, quant_conv, 1, nullptr, other(), nullptr, h.H, h.W); flip();

  // ---- vector quantize (z is [P, 256] row-contiguous in NHWC) ----
  {
    long P = (long)h.B * h.H * h.W;                      // 8192
    k_vq_argmin<<<(int)(P / 16), 32, 0, stream>>>(h.p, cbbf, cbn, vidx, 8192);
    bf16* zq = other();
    k_vq_gather<<<ngrid(P * 32, 256), 256, 0, stream>>>(vidx, cbbf, zq, P);
    h.p = zq; h.C = 256; flip();
  }

  // ---- decoder ----
  h = runConv(h, post_quant,  1, nullptr, other(), nullptr, h.H, h.W); flip();
  h = runConv(h, dec_conv_in, 1, nullptr, other(), nullptr, h.H, h.W); flip();
  res(dec_mid.b1); attn(dec_mid.attn); res(dec_mid.b2);
  res(dU2b[0]); res(dU2b[1]); res(dU2b[2]); up(dU2up);
  res(dU1b[0]); res(dU1b[1]); res(dU1b[2]); up(dU1up);
  res(dU0b[0]); res(dU0b[1]); res(dU0b[2]);
  h = runGN(h, dec_norm_out, 1, Bb[2]);
  runConv(h, dec_conv_out, 1, nullptr, nullptr, (float*)d_out, h.H, h.W);
}
